// SimpleRNN_32375463477377
// MI455X (gfx1250) — compile-verified, matrix operand order checked by reading, not run on hardware
//
#include <hip/hip_runtime.h>

typedef __attribute__((ext_vector_type(16))) __bf16 v16bf;
typedef __attribute__((ext_vector_type(8)))  __bf16 v8bf;
typedef __attribute__((ext_vector_type(8)))  float  v8f;

#define RNN_VOCAB  32000
#define RNN_EMBED  512
#define RNN_HIDDEN 1024
#define RNN_KTOT   1536   // HIDDEN + EMBED
#define RNN_N      8192
#define RNN_L      16

// ---------- helpers ----------
__device__ __forceinline__ __bf16 f2bf(float f) {
    union { float f; unsigned u; } in; in.f = f;
    unsigned u = in.u;
    unsigned r = u + 0x7FFFu + ((u >> 16) & 1u);   // round-to-nearest-even
    union { unsigned short s; __bf16 b; } out;
    out.s = (unsigned short)(r >> 16);
    return out.b;
}

// Hardware tanh (single TRANS32 v_tanh_f32 on CDNA5) with safe fallback.
__device__ __forceinline__ float fast_tanh(float x) {
#if __has_builtin(__builtin_amdgcn_tanhf)
    return __builtin_amdgcn_tanhf(x);
#else
    return tanhf(x);
#endif
}

// Load one 16x32 bf16 WMMA operand fragment for this lane.
// rowp: per-lane pointer to the start of this lane's (row-major) row.
// k0:   K offset (multiple of 32). half = lane>>4 selects the K sub-chunks:
//   half=0 -> K[k0..k0+7] and K[k0+16..k0+23]
//   half=1 -> K[k0+8..k0+15] and K[k0+24..k0+31]
__device__ __forceinline__ v16bf load_frag(const __bf16* rowp, int k0, int half) {
    const v8bf lo = *(const v8bf*)(rowp + k0 + half * 8);
    const v8bf hi = *(const v8bf*)(rowp + k0 + half * 8 + 16);
    v16bf a;
#pragma unroll
    for (int i = 0; i < 8; ++i) { a[i] = lo[i]; a[8 + i] = hi[i]; }
    return a;
}

__device__ __forceinline__ v8f wmma_bf16(v16bf a, v16bf b, v8f c) {
    return __builtin_amdgcn_wmma_f32_16x16x32_bf16(false, a, false, b, (short)0, c, false, false);
}

// ---------- prep kernels ----------
__global__ void rnn_cvt_bf16(const float* __restrict__ src, __bf16* __restrict__ dst, int n) {
    int g = blockIdx.x * blockDim.x + threadIdx.x;
    if (g < n) dst[g] = f2bf(src[g]);
}

__global__ void rnn_build_wc(const float* __restrict__ Whh, const float* __restrict__ Wxh,
                             __bf16* __restrict__ Wc) {
    int g = blockIdx.x * blockDim.x + threadIdx.x;
    if (g >= RNN_HIDDEN * RNN_KTOT) return;
    int h = g / RNN_KTOT;
    int k = g % RNN_KTOT;
    float v = (k < RNN_HIDDEN) ? Whh[h * RNN_HIDDEN + k]
                               : Wxh[h * RNN_EMBED + (k - RNN_HIDDEN)];
    Wc[g] = f2bf(v);
}

__global__ void rnn_build_bias(const float* __restrict__ bhh, const float* __restrict__ bxh,
                               float* __restrict__ bs) {
    int g = blockIdx.x * blockDim.x + threadIdx.x;
    if (g < RNN_HIDDEN) bs[g] = bhh[g] + bxh[g];
}

__global__ void rnn_zero_bf16(__bf16* __restrict__ p, int n) {
    int g = blockIdx.x * blockDim.x + threadIdx.x;
    if (g < n) p[g] = f2bf(0.0f);
}

// ---------- recurrent step: H_out = tanh([H_in | E[X_t]] @ Wc^T + bias) ----------
// grid: (64, 8)  block: 256 (8 waves). Wave tile = 32(M) x 64(N).
// Block tile = 128(M) x 128(N). 8 WMMAs per 12 B128-load K-iteration.
__global__ __launch_bounds__(256)
void rnn_step(const __bf16* __restrict__ Hin,
              __bf16* __restrict__ Hout,
              const __bf16* __restrict__ Ebf,
              const __bf16* __restrict__ Wc,
              const float*  __restrict__ biasSum,
              const int*    __restrict__ X,
              int t,
              float* __restrict__ outF)   // non-null only on the last step
{
    const int tid  = threadIdx.x;
    const int lane = tid & 31;
    const int wid  = tid >> 5;          // 0..7
    const int wm   = wid & 3;           // 4 waves along M
    const int wn   = wid >> 2;          // 2 waves along N
    const int m0   = blockIdx.x * 128 + wm * 32;
    const int n0   = blockIdx.y * 128 + wn * 64;

    const int r    = lane & 15;
    const int half = lane >> 4;

    // Per-lane row pointers (A layout: M = lane%16 for all lanes).
    const __bf16* pa0 = Hin + (size_t)(m0 + r)      * RNN_HIDDEN;
    const __bf16* pa1 = Hin + (size_t)(m0 + 16 + r) * RNN_HIDDEN;
    const __bf16* pb[4];
#pragma unroll
    for (int j = 0; j < 4; ++j)
        pb[j] = Wc + (size_t)(n0 + j * 16 + r) * RNN_KTOT;

    const int idx0 = X[(size_t)(m0 + r)      * RNN_L + t];
    const int idx1 = X[(size_t)(m0 + 16 + r) * RNN_L + t];
    const __bf16* pe0 = Ebf + (size_t)idx0 * RNN_EMBED;
    const __bf16* pe1 = Ebf + (size_t)idx1 * RNN_EMBED;

    v8f c0[4] = {};   // rows m0..m0+15   x 4 column tiles
    v8f c1[4] = {};   // rows m0+16..+31 x 4 column tiles

    // K part 1: recurrent term, K = 0..1024 over Hin
#pragma unroll 2
    for (int k = 0; k < RNN_HIDDEN; k += 32) {
        v16bf a0 = load_frag(pa0, k, half);
        v16bf a1 = load_frag(pa1, k, half);
        v16bf b[4];
#pragma unroll
        for (int j = 0; j < 4; ++j) b[j] = load_frag(pb[j], k, half);
#pragma unroll
        for (int j = 0; j < 4; ++j) {
            c0[j] = wmma_bf16(a0, b[j], c0[j]);
            c1[j] = wmma_bf16(a1, b[j], c1[j]);
        }
    }

    // K part 2: input projection via embedding gather, K = 1024..1536
#pragma unroll 2
    for (int k = 0; k < RNN_EMBED; k += 32) {
        v16bf a0 = load_frag(pe0, k, half);
        v16bf a1 = load_frag(pe1, k, half);
        v16bf b[4];
#pragma unroll
        for (int j = 0; j < 4; ++j) b[j] = load_frag(pb[j], RNN_HIDDEN + k, half);
#pragma unroll
        for (int j = 0; j < 4; ++j) {
            c0[j] = wmma_bf16(a0, b[j], c0[j]);
            c1[j] = wmma_bf16(a1, b[j], c1[j]);
        }
    }

    // Epilogue. C/D layout: lane holds column n = lane%16; VGPR i holds row
    // m = i (lanes 0-15) / i+8 (lanes 16-31).
    int   nCol[4];
    float bsv[4];
#pragma unroll
    for (int j = 0; j < 4; ++j) {
        nCol[j] = n0 + j * 16 + r;
        bsv[j]  = biasSum[nCol[j]];
    }

#pragma unroll
    for (int i = 0; i < 8; ++i) {
        const int mr0 = m0 + i + half * 8;
        const int mr1 = m0 + 16 + i + half * 8;
#pragma unroll
        for (int j = 0; j < 4; ++j) {
            const float v0 = fast_tanh(c0[j][i] + bsv[j]);
            const float v1 = fast_tanh(c1[j][i] + bsv[j]);
            if (outF) {   // last step: only the f32 result is live
                outF[(size_t)mr0 * RNN_HIDDEN + nCol[j]] = v0;
                outF[(size_t)mr1 * RNN_HIDDEN + nCol[j]] = v1;
            } else {      // intermediate step: bf16 state for the next GEMM
                Hout[(size_t)mr0 * RNN_HIDDEN + nCol[j]] = f2bf(v0);
                Hout[(size_t)mr1 * RNN_HIDDEN + nCol[j]] = f2bf(v1);
            }
        }
    }
}

// ---------- launcher ----------
extern "C" void kernel_launch(void* const* d_in, const int* in_sizes, int n_in,
                              void* d_out, int out_size, void* d_ws, size_t ws_size,
                              hipStream_t stream) {
    const int*   X   = (const int*)d_in[0];
    const float* E   = (const float*)d_in[1];
    const float* Whh = (const float*)d_in[2];
    const float* bhh = (const float*)d_in[3];
    const float* Wxh = (const float*)d_in[4];
    const float* bxh = (const float*)d_in[5];
    float* out = (float*)d_out;

    char* ws = (char*)d_ws;
    size_t off = 0;
    auto alloc = [&](size_t bytes) -> void* {
        void* p = ws + off;
        off = (off + bytes + 255) & ~(size_t)255;
        return p;
    };

    __bf16* Ebf = (__bf16*)alloc((size_t)RNN_VOCAB * RNN_EMBED * sizeof(__bf16));
    __bf16* Wc  = (__bf16*)alloc((size_t)RNN_HIDDEN * RNN_KTOT * sizeof(__bf16));
    float*  bs  = (float*)alloc((size_t)RNN_HIDDEN * sizeof(float));
    __bf16* H0  = (__bf16*)alloc((size_t)RNN_N * RNN_HIDDEN * sizeof(__bf16));
    __bf16* H1  = (__bf16*)alloc((size_t)RNN_N * RNN_HIDDEN * sizeof(__bf16));
    (void)ws_size; (void)in_sizes; (void)n_in; (void)out_size;

    // Prep: convert E, pack [W_hh | W_xh] row-major bf16, bias sum, zero H0.
    {
        int n = RNN_VOCAB * RNN_EMBED;
        rnn_cvt_bf16<<<(n + 255) / 256, 256, 0, stream>>>(E, Ebf, n);
    }
    {
        int n = RNN_HIDDEN * RNN_KTOT;
        rnn_build_wc<<<(n + 255) / 256, 256, 0, stream>>>(Whh, Wxh, Wc);
    }
    rnn_build_bias<<<(RNN_HIDDEN + 255) / 256, 256, 0, stream>>>(bhh, bxh, bs);
    {
        int n = RNN_N * RNN_HIDDEN;
        rnn_zero_bf16<<<(n + 255) / 256, 256, 0, stream>>>(H0, n);
    }

    // 16 recurrent steps, double-buffered H; last step also writes f32 output.
    dim3 grid(RNN_N / 128, RNN_HIDDEN / 128);
    dim3 block(256);
    __bf16* Hbuf[2] = { H0, H1 };
    for (int t = 0; t < RNN_L; ++t) {
        const __bf16* Hin  = Hbuf[t & 1];
        __bf16*       Hout = Hbuf[(t + 1) & 1];
        float* outF = (t == RNN_L - 1) ? out : nullptr;
        rnn_step<<<grid, block, 0, stream>>>(Hin, Hout, Ebf, Wc, bs, X, t, outF);
    }
}
